// SurpriseGatedModel_35270271435257
// MI455X (gfx1250) — matrix operand order repaired and checked by reading, not compiled, not run on hardware
//
#include <hip/hip_runtime.h>

// ---------------------------------------------------------------------------
// SurpriseGatedModel for MI455X (gfx1250, wave32)
//   K1: WMMA-based encoder (embed gather -> 32x64 GEMM -> relu -> 64x32 GEMM
//       -> residual -> LayerNorm), 16 tokens per wave, 8 v_wmma per tile.
//   K2: fast-weight scan, one wave per batch item, lane = hidden dim,
//       fold-reduce all-reduces (readlane broadcast), software-pipelined
//       k/v loads, wave-uniform surprise gate.
//   K3: logits head + write-rate reduction.
// ---------------------------------------------------------------------------

typedef __attribute__((ext_vector_type(16))) _Float16 v16h;
typedef __attribute__((ext_vector_type(8)))  float    v8f;

#define HID    32
#define INNER  64      // 2*HIDDEN_DIM
#define MLP    8
#define VOCAB  64
#define BSZ    256
#define LSEQ   4096
#define NPAIR  2047    // (L - 3 + 1) / 2
#define LN_EPSF 1e-5f

// ===========================================================================
// Kernel 1: encoder.  One wave handles a 16-token tile; 4 waves / block.
// ===========================================================================
__global__ __launch_bounds__(128) void encoder_kernel(
    const int*   __restrict__ seq,
    const float* __restrict__ embed,
    const float* __restrict__ ff1_w,   // (64,32)  z = h @ ff1_w^T
    const float* __restrict__ ff1_b,
    const float* __restrict__ ff2_w,   // (32,64)  f = z @ ff2_w^T
    const float* __restrict__ ff2_b,
    const float* __restrict__ ln_g,
    const float* __restrict__ ln_b,
    float*       __restrict__ hidden)  // (B*L, 32)
{
    __shared__ float    lds_h[4][16][HID];    // per-wave h tile (residual)
    __shared__ _Float16 lds_z[4][16][INNER];  // per-wave z tile (relayout)

    const int tid  = threadIdx.x;
    const int wave = tid >> 5;
    const int lane = tid & 31;
    const int r    = lane & 15;   // tile row (M) / N column, depending on use
    const int hi   = lane >> 4;   // half-wave select

    // ---- stage weights in VGPRs as f16 (8 KB total, L2-resident) ----------
    // GEMM1 B tiles: B1[nt][K=h][n] = ff1_w[nt*16+n][h]
    // assumed B layout: lane = n + 16*khalf holds K = {khalf*8 + j, 16 + khalf*8 + j}
    v16h B1[4];
#pragma unroll
    for (int nt = 0; nt < 4; ++nt) {
        const float* src = ff1_w + (nt * 16 + r) * HID;
#pragma unroll
        for (int j = 0; j < 8; ++j) { B1[nt][j] = (_Float16)src[hi * 8 + j]; B1[nt][8 + j] = (_Float16)src[16 + hi * 8 + j]; }
    }
    // GEMM2 B tiles: B2[nt][kt][K=k2][n] = ff2_w[nt*16+n][kt*32+k2]
    v16h B2[2][2];
#pragma unroll
    for (int nt = 0; nt < 2; ++nt)
#pragma unroll
        for (int kt = 0; kt < 2; ++kt) {
            const float* src = ff2_w + (nt * 16 + r) * INNER + kt * 32;
#pragma unroll
            for (int j = 0; j < 8; ++j) { B2[nt][kt][j] = (_Float16)src[hi * 8 + j]; B2[nt][kt][8 + j] = (_Float16)src[16 + hi * 8 + j]; }
        }
    float bias1[4];
#pragma unroll
    for (int nt = 0; nt < 4; ++nt) bias1[nt] = ff1_b[nt * 16 + r];
    float bias2[2], g_ln[2], b_ln[2];
#pragma unroll
    for (int nt = 0; nt < 2; ++nt) {
        bias2[nt] = ff2_b[nt * 16 + r];
        g_ln[nt]  = ln_g[nt * 16 + r];
        b_ln[nt]  = ln_b[nt * 16 + r];
    }

    const long tile = (long)blockIdx.x * 4 + wave;
    const long t0   = tile * 16;

    // ---- gather embedding row into WMMA A-layout (f16) + LDS (f32) --------
    // A 16x32 f16 layout: lo lanes K in {0..7, 16..23}; hi lanes {8..15, 24..31}
    const int   tok  = seq[t0 + r];
    const float* erow = embed + (long)tok * HID;
    const int k0 = hi * 8, k1 = 16 + hi * 8;
    v16h A;
#pragma unroll
    for (int j = 0; j < 8; ++j) {
        const float f0 = erow[k0 + j];
        const float f1 = erow[k1 + j];
        A[j]     = (_Float16)f0;
        A[8 + j] = (_Float16)f1;
        lds_h[wave][r][k0 + j] = f0;
        lds_h[wave][r][k1 + j] = f1;
    }
    __syncthreads();

    // ---- GEMM1: z(16x64) = A(16x32) x B1 ----------------------------------
    v8f zacc[4];
#pragma unroll
    for (int nt = 0; nt < 4; ++nt) {
        v8f c = {};
        zacc[nt] = __builtin_amdgcn_wmma_f32_16x16x32_f16(
            false, A, false, B1[nt], (short)0, c, false, false);
    }
    // bias + relu, spill to LDS row-major as f16 for the A-relayout
    // C layout: element v -> M = v + 8*hi, N = r (within tile nt)
#pragma unroll
    for (int nt = 0; nt < 4; ++nt) {
        const int n = nt * 16 + r;
#pragma unroll
        for (int v = 0; v < 8; ++v) {
            float z = zacc[nt][v] + bias1[nt];
            z = z > 0.f ? z : 0.f;
            lds_z[wave][v + 8 * hi][n] = (_Float16)z;
        }
    }
    __syncthreads();

    // ---- reload z in A-layout (two K=32 tiles) -----------------------------
    v16h ZA[2];
#pragma unroll
    for (int kt = 0; kt < 2; ++kt) {
        const _Float16* zr = &lds_z[wave][r][kt * 32];
#pragma unroll
        for (int j = 0; j < 8; ++j) {
            ZA[kt][j]     = zr[hi * 8 + j];
            ZA[kt][8 + j] = zr[16 + hi * 8 + j];
        }
    }

    // ---- GEMM2: f(16x32) = z(16x64) x B2, K accumulated over 2 WMMAs ------
    v8f x0, x1;
    {
        v8f c = {};
        c  = __builtin_amdgcn_wmma_f32_16x16x32_f16(false, ZA[0], false, B2[0][0], (short)0, c, false, false);
        x0 = __builtin_amdgcn_wmma_f32_16x16x32_f16(false, ZA[1], false, B2[0][1], (short)0, c, false, false);
        v8f d = {};
        d  = __builtin_amdgcn_wmma_f32_16x16x32_f16(false, ZA[0], false, B2[1][0], (short)0, d, false, false);
        x1 = __builtin_amdgcn_wmma_f32_16x16x32_f16(false, ZA[1], false, B2[1][1], (short)0, d, false, false);
    }

    // ---- bias + residual + LayerNorm over 32 dims --------------------------
    const int n0 = r, n1 = 16 + r;
#pragma unroll
    for (int v = 0; v < 8; ++v) {
        const int m = v + 8 * hi;
        x0[v] += bias2[0] + lds_h[wave][m][n0];
        x1[v] += bias2[1] + lds_h[wave][m][n1];
    }
    // row m is held by the 16 lanes sharing 'hi' -> 4-stage butterfly
#pragma unroll
    for (int v = 0; v < 8; ++v) {
        float s = x0[v] + x1[v];
#pragma unroll
        for (int msk = 1; msk <= 8; msk <<= 1) s += __shfl_xor(s, msk, 32);
        const float mu = s * (1.f / 32.f);
        const float e0 = x0[v] - mu, e1 = x1[v] - mu;
        float q = e0 * e0 + e1 * e1;
#pragma unroll
        for (int msk = 1; msk <= 8; msk <<= 1) q += __shfl_xor(q, msk, 32);
        const float rs = rsqrtf(q * (1.f / 32.f) + LN_EPSF);
        x0[v] = g_ln[0] * e0 * rs + b_ln[0];
        x1[v] = g_ln[1] * e1 * rs + b_ln[1];
    }

    // ---- bounce through LDS for coalesced float4 stores --------------------
    __syncthreads();
#pragma unroll
    for (int v = 0; v < 8; ++v) {
        const int m = v + 8 * hi;
        lds_h[wave][m][n0] = x0[v];
        lds_h[wave][m][n1] = x1[v];
    }
    __syncthreads();
    float*       orow = hidden + (t0 + r) * HID + hi * 16;
    const float* srow = &lds_h[wave][r][hi * 16];
    const float4* s4 = (const float4*)srow;
    float4*       o4 = (float4*)orow;
#pragma unroll
    for (int j = 0; j < 4; ++j) o4[j] = s4[j];
}

// ===========================================================================
// Fold-reduce: sum p[0..7] over all 32 lanes, result broadcast to out[0..7].
// Stages over masks 1,2,4 halve the vector (keep/forward complementary
// halves), leaving lane l of each octet with the octet-sum of output
// j(l) = 4*(l&1) + 2*((l>>1)&1) + ((l>>2)&1)  (3-bit reversal).
// Masks 8,16 finish the 32-lane sum; constant-lane __shfl (-> v_readlane)
// broadcasts the 8 results.  ~9 ds-swizzles + 8 readlanes vs 40 swizzles.
// ===========================================================================
__device__ __forceinline__ void allreduce8(const float* __restrict__ p,
                                           float* __restrict__ out,
                                           const int lane)
{
    float q[4];
#pragma unroll
    for (int j = 0; j < 4; ++j) {
        const float keep = (lane & 1) ? p[j + 4] : p[j];
        const float send = (lane & 1) ? p[j]     : p[j + 4];
        q[j] = keep + __shfl_xor(send, 1, 32);
    }
    float r2[2];
#pragma unroll
    for (int j = 0; j < 2; ++j) {
        const float keep = (lane & 2) ? q[j + 2] : q[j];
        const float send = (lane & 2) ? q[j]     : q[j + 2];
        r2[j] = keep + __shfl_xor(send, 2, 32);
    }
    float s;
    {
        const float keep = (lane & 4) ? r2[1] : r2[0];
        const float send = (lane & 4) ? r2[0] : r2[1];
        s = keep + __shfl_xor(send, 4, 32);
    }
    s += __shfl_xor(s, 8, 32);
    s += __shfl_xor(s, 16, 32);
    // S[j] lives in lane brev3(j): 0,4,2,6,1,5,3,7
    out[0] = __shfl(s, 0, 32);
    out[1] = __shfl(s, 4, 32);
    out[2] = __shfl(s, 2, 32);
    out[3] = __shfl(s, 6, 32);
    out[4] = __shfl(s, 1, 32);
    out[5] = __shfl(s, 5, 32);
    out[6] = __shfl(s, 3, 32);
    out[7] = __shfl(s, 7, 32);
}

// ===========================================================================
// Kernel 2: fast-weight scan.  One wave per batch item; lane = hidden dim.
// ===========================================================================
__global__ __launch_bounds__(32) void scan_kernel(
    const float* __restrict__ hidden,
    const float* __restrict__ fc1_w,   // (8,32)
    const float* __restrict__ fc1_b,   // (8,)
    const float* __restrict__ fc2_w,   // (32,8)
    const float* __restrict__ fc2_b,   // (32,)
    float*       __restrict__ ctx_out, // (B,32)
    float*       __restrict__ wcnt_out)// (B,)
{
    const int b = blockIdx.x;
    const int h = threadIdx.x;                 // 0..31 == hidden dim
    const float* base = hidden + (long)b * LSEQ * HID;

    float W1c[MLP], W2r[MLP], b1[MLP];
#pragma unroll
    for (int j = 0; j < MLP; ++j) {
        W1c[j] = fc1_w[j * HID + h];           // column h of W1
        W2r[j] = fc2_w[h * MLP + j];           // row h of W2
        b1[j]  = fc1_b[j];                     // replicated
    }
    float b2     = fc2_b[h];
    float ssum   = 0.f;
    float cnt    = 0.f;
    float writes = 0.f;

    // software pipeline: loads for step i+1 issue before step i's math
    float kh = base[h];
    float vh = base[HID + h];

    for (int i = 0; i < NPAIR; ++i) {
        float kh_n = 0.f, vh_n = 0.f;
        if (i + 1 < NPAIR) {
            const float* kp = base + (long)(2 * (i + 1)) * HID;
            kh_n = kp[h];
            vh_n = kp[HID + h];
        }
        if (i + 8 < NPAIR)
            __builtin_prefetch(base + (long)(2 * (i + 8)) * HID + h, 0, 1);

        // surprise gate: dist^2 vs THRESHOLD^2 (wave-uniform after reduce)
        const float mean = ssum / fmaxf(cnt, 1.f);
        float d2 = (kh - mean) * (kh - mean);
#pragma unroll
        for (int msk = 1; msk < 32; msk <<= 1) d2 += __shfl_xor(d2, msk, 32);
        const bool write = (cnt == 0.f) || (d2 > 16.f);

        // z1 = W1 k + b1
        float p[MLP], z1[MLP];
#pragma unroll
        for (int j = 0; j < MLP; ++j) p[j] = W1c[j] * kh;
        allreduce8(p, z1, h);
        float a[MLP];
#pragma unroll
        for (int j = 0; j < MLP; ++j) { z1[j] += b1[j]; a[j] = fmaxf(z1[j], 0.f); }

        // pred[h] = W2[h,:] a + b2[h]   (lane-local)
        float pred = b2;
#pragma unroll
        for (int j = 0; j < MLP; ++j) pred += W2r[j] * a[j];
        const float dpred = (2.f / (float)HID) * (pred - vh);

        // dz1 = (W2^T dpred) * (z1 > 0)
        float dz[MLP];
#pragma unroll
        for (int j = 0; j < MLP; ++j) p[j] = W2r[j] * dpred;
        allreduce8(p, dz, h);
#pragma unroll
        for (int j = 0; j < MLP; ++j) dz[j] = (z1[j] > 0.f) ? dz[j] : 0.f;

        if (write) {                            // wave-uniform branch
#pragma unroll
            for (int j = 0; j < MLP; ++j) {
                W1c[j] -= 0.01f * dz[j] * kh;   // dW1 = outer(dz1, k)
                b1[j]  -= 0.01f * dz[j];
                W2r[j] -= 0.01f * dpred * a[j]; // dW2 = outer(dpred, a)
            }
            b2 -= 0.01f * dpred;
            writes += 1.f;
        }
        ssum += kh;
        cnt  += 1.f;
        kh = kh_n;
        vh = vh_n;
    }

    // context = W2 relu(W1 q + b1) + b2
    const float qh = base[(long)(LSEQ - 1) * HID + h];
    float p[MLP], z1[MLP];
#pragma unroll
    for (int j = 0; j < MLP; ++j) p[j] = W1c[j] * qh;
    allreduce8(p, z1, h);
    float ctx = b2;
#pragma unroll
    for (int j = 0; j < MLP; ++j) ctx += W2r[j] * fmaxf(z1[j] + b1[j], 0.f);

    ctx_out[b * HID + h] = ctx;
    if (h == 0) wcnt_out[b] = writes;
}

// ===========================================================================
// Kernel 3: logits head + write-rate reduction.
// ===========================================================================
__global__ __launch_bounds__(64) void head_kernel(
    const float* __restrict__ ctx,
    const float* __restrict__ wcnt,
    const float* __restrict__ out_w,   // (64,32)
    const float* __restrict__ out_b,   // (64,)
    float*       __restrict__ out)     // 16384 logits + 1 write_rate
{
    const int blk = blockIdx.x;
    if (blk < BSZ) {
        const int o = threadIdx.x;
        const float* c = ctx + blk * HID;
        const float* w = out_w + o * HID;
        float s = out_b[o];
#pragma unroll
        for (int j = 0; j < HID; ++j) s += c[j] * w[j];
        out[blk * VOCAB + o] = s;
    } else {
        __shared__ float red[64];
        float s = 0.f;
        for (int j = threadIdx.x; j < BSZ; j += 64) s += wcnt[j];
        red[threadIdx.x] = s;
        __syncthreads();
        for (int st = 32; st >= 1; st >>= 1) {
            if (threadIdx.x < st) red[threadIdx.x] += red[threadIdx.x + st];
            __syncthreads();
        }
        if (threadIdx.x == 0)
            out[BSZ * VOCAB] = red[0] / (float)(BSZ * NPAIR);
    }
}

// ===========================================================================
extern "C" void kernel_launch(void* const* d_in, const int* in_sizes, int n_in,
                              void* d_out, int out_size, void* d_ws, size_t ws_size,
                              hipStream_t stream) {
    const int*   seq   = (const int*)  d_in[0];
    const float* embed = (const float*)d_in[1];
    const float* ff1_w = (const float*)d_in[2];
    const float* ff1_b = (const float*)d_in[3];
    const float* ff2_w = (const float*)d_in[4];
    const float* ff2_b = (const float*)d_in[5];
    const float* ln_g  = (const float*)d_in[6];
    const float* ln_b  = (const float*)d_in[7];
    const float* fc1_w = (const float*)d_in[8];
    const float* fc1_b = (const float*)d_in[9];
    const float* fc2_w = (const float*)d_in[10];
    const float* fc2_b = (const float*)d_in[11];
    const float* out_w = (const float*)d_in[12];
    const float* out_b = (const float*)d_in[13];

    // workspace layout: hidden (128 MB) | ctx (B*32) | write counts (B)
    float* hidden = (float*)d_ws;
    float* ctx    = hidden + (size_t)BSZ * LSEQ * HID;
    float* wcnt   = ctx + (size_t)BSZ * HID;

    // encoder: 65536 16-token tiles, 4 waves (tiles) per 128-thread block
    encoder_kernel<<<(BSZ * LSEQ) / 64, 128, 0, stream>>>(
        seq, embed, ff1_w, ff1_b, ff2_w, ff2_b, ln_g, ln_b, hidden);

    // scan: one wave per batch item
    scan_kernel<<<BSZ, 32, 0, stream>>>(
        hidden, fc1_w, fc1_b, fc2_w, fc2_b, ctx, wcnt);

    // head: 256 logit blocks + 1 reduction block
    head_kernel<<<BSZ + 1, 64, 0, stream>>>(
        ctx, wcnt, out_w, out_b, (float*)d_out);
}
